// PolarLinear_40492951667458
// MI455X (gfx1250) — compile-verified, matrix operand order checked by reading, not run on hardware
//
#include <hip/hip_runtime.h>
#include <hip/hip_bf16.h>

typedef __attribute__((ext_vector_type(16))) _Float16 v16h;
typedef __attribute__((ext_vector_type(8)))  _Float16 v8h;
typedef __attribute__((ext_vector_type(8)))  float    v8f;

#define IN_F   4096
#define OUT_F  11008
#define M_TOK  2048           // 4 * 512 tokens
#define BS     128            // hadamard block
#define INV_SQRT_BS 0.08838834764831845f  // 1/sqrt(128)

// ---------------------------------------------------------------------------
// Kernel 1: fast Walsh-Hadamard transform of x (per token, per 128-chunk),
// normalized (x -> H x with H = normalized Sylvester Hadamard), emit fp16.
// ---------------------------------------------------------------------------
__global__ __launch_bounds__(256) void fwht_x_kernel(const float* __restrict__ x,
                                                     _Float16* __restrict__ xh) {
    __shared__ float buf[2][BS];
    const int chunk = threadIdx.x >> 7;       // 0..1 (two 128-chunks per block)
    const int t     = threadIdx.x & 127;
    const long base = (long)blockIdx.x * 256 + (long)chunk * BS + t;

    buf[chunk][t] = x[base];
    __syncthreads();
#pragma unroll
    for (int s = 0; s < 7; ++s) {
        const int bit = 1 << s;
        const float a = buf[chunk][t ^ bit];
        const float b = buf[chunk][t];
        __syncthreads();
        buf[chunk][t] = (t & bit) ? (a - b) : (b + a);
        __syncthreads();
    }
    xh[base] = (_Float16)(buf[chunk][t] * INV_SQRT_BS);
}

// ---------------------------------------------------------------------------
// Kernel 2: decode 4-bit codes -> fp16 W' (no Hadamard; folded into x).
// w'[o,i] = centroids[code] * (1/sqrt(128)) * norms[o*32 + i/128]
// ---------------------------------------------------------------------------
__global__ __launch_bounds__(256) void decode_w_kernel(const int* __restrict__ codes,
                                                       const _Float16* __restrict__ norms,
                                                       const float* __restrict__ centroids,
                                                       _Float16* __restrict__ wq) {
    __shared__ float cent[16];
    if (threadIdx.x < 16)
        cent[threadIdx.x] = centroids[threadIdx.x] * INV_SQRT_BS;
    __syncthreads();

    const long base = ((long)blockIdx.x * 256 + threadIdx.x) * 8;
    const float nrm = (float)norms[base >> 7];

    const int4 c0 = ((const int4*)(codes + base))[0];
    const int4 c1 = ((const int4*)(codes + base))[1];

    v8h o;
    o[0] = (_Float16)(cent[c0.x] * nrm);
    o[1] = (_Float16)(cent[c0.y] * nrm);
    o[2] = (_Float16)(cent[c0.z] * nrm);
    o[3] = (_Float16)(cent[c0.w] * nrm);
    o[4] = (_Float16)(cent[c1.x] * nrm);
    o[5] = (_Float16)(cent[c1.y] * nrm);
    o[6] = (_Float16)(cent[c1.z] * nrm);
    o[7] = (_Float16)(cent[c1.w] * nrm);
    *(v8h*)(wq + base) = o;
}

// ---------------------------------------------------------------------------
// Kernel 3: WMMA GEMM  out[M, N] = xh[M, K] * wq[N, K]^T   (fp16 in, fp32 acc)
// Block tile 128x256, BK=32, double-buffered LDS (one barrier per K-step).
// 8 waves as (mw 0..1) x (nw 0..3); each wave owns a 64x64 = 4x4 WMMA tile.
// All global accesses: uniform base pointer + 32-bit per-thread offset
// (+ uniform kb*BK term) so address-space inference emits global_load_b128,
// not flat_load_b128 (flat would tie staging loads into DScnt with the
// WMMA fragment ds_loads).
// ---------------------------------------------------------------------------
#define BM    128
#define BN    256
#define BK    32
#define LSTR  40            // LDS row stride in halves: 80 B, 16B-aligned, conflict-free
#define NKIT  (IN_F / BK)   // 128

__global__ __launch_bounds__(256) void gemm_wmma_kernel(const _Float16* __restrict__ A, // xh [M, K]
                                                        const _Float16* __restrict__ B, // wq [N, K]
                                                        _Float16* __restrict__ out) {
    __shared__ _Float16 sA[2][BM * LSTR];   // 2 x 10 KB
    __shared__ _Float16 sB[2][BN * LSTR];   // 2 x 20 KB

    const int nBase = blockIdx.x * BN;      // 43 tiles
    const int mBase = blockIdx.y * BM;      // 16 tiles
    const int tid   = threadIdx.x;
    const int wave  = tid >> 5;
    const int lane  = tid & 31;
    const int hsel  = lane >> 4;            // which K-half of the fragment
    const int r     = lane & 15;            // row-within-tile / N-within-tile
    const int mw    = wave & 1;             // 64-row slab
    const int nw    = wave >> 1;            // 64-col slab

    // uniform (scalar) block bases; per-thread offsets stay 32-bit
    const _Float16* __restrict__ Ab = A + (long)mBase * IN_F;
    const _Float16* __restrict__ Bb = B + (long)nBase * IN_F;

    // ---- staging maps: rows of 32 halves = 4 x 16B chunks per row
    // A tile: 128*4 = 512 chunks -> 2/thread; B tile: 256*4 = 1024 -> 4/thread
    int aoffL[2], boffL[4];                 // LDS half-offsets
    int aoffG[2], boffG[4];                 // global element offsets (32-bit)
#pragma unroll
    for (int i = 0; i < 2; ++i) {
        const int c   = tid + i * 256;
        const int row = c >> 2, col = (c & 3) * 8;
        aoffL[i] = row * LSTR + col;
        aoffG[i] = row * IN_F + col;
    }
#pragma unroll
    for (int i = 0; i < 4; ++i) {
        const int c   = tid + i * 256;
        const int row = c >> 2, col = (c & 3) * 8;
        boffL[i] = row * LSTR + col;
        boffG[i] = row * IN_F + col;
    }

    v8f acc[4][4];
#pragma unroll
    for (int mt = 0; mt < 4; ++mt)
#pragma unroll
        for (int nt = 0; nt < 4; ++nt)
            acc[mt][nt] = (v8f){};

    // ---- preload K-step 0 into buffer 0
    {
        v8h ra[2], rb[4];
#pragma unroll
        for (int i = 0; i < 2; ++i) ra[i] = *(const v8h*)(Ab + aoffG[i]);
#pragma unroll
        for (int i = 0; i < 4; ++i) rb[i] = *(const v8h*)(Bb + boffG[i]);
#pragma unroll
        for (int i = 0; i < 2; ++i) *(v8h*)&sA[0][aoffL[i]] = ra[i];
#pragma unroll
        for (int i = 0; i < 4; ++i) *(v8h*)&sB[0][boffL[i]] = rb[i];
    }
    __syncthreads();

    int cur = 0;
    for (int kb = 0; kb < NKIT; ++kb) {
        // ---- issue next K-step's global loads (overlap with WMMA below)
        v8h ra[2], rb[4];
        const bool more = (kb + 1 < NKIT);
        if (more) {
            const int kG = (kb + 1) * BK;   // uniform
#pragma unroll
            for (int i = 0; i < 2; ++i) ra[i] = *(const v8h*)(Ab + kG + aoffG[i]);
#pragma unroll
            for (int i = 0; i < 4; ++i) rb[i] = *(const v8h*)(Bb + kG + boffG[i]);
        }

        // ---- fragments from LDS buffer `cur`
        v16h af[4], bf[4];
#pragma unroll
        for (int mt = 0; mt < 4; ++mt) {
            const _Float16* p = &sA[cur][(mw * 64 + mt * 16 + r) * LSTR + hsel * 8];
            const v8h lo = *(const v8h*)p;          // K = hsel*8 .. +7
            const v8h hi = *(const v8h*)(p + 16);   // K = 16 + hsel*8 .. +7
            v16h a;
#pragma unroll
            for (int i = 0; i < 8; ++i) { a[i] = lo[i]; a[i + 8] = hi[i]; }
            af[mt] = a;
        }
#pragma unroll
        for (int nt = 0; nt < 4; ++nt) {
            const _Float16* p = &sB[cur][(nw * 64 + nt * 16 + r) * LSTR + hsel * 8];
            const v8h lo = *(const v8h*)p;
            const v8h hi = *(const v8h*)(p + 16);
            v16h b;
#pragma unroll
            for (int i = 0; i < 8; ++i) { b[i] = lo[i]; b[i + 8] = hi[i]; }
            bf[nt] = b;
        }

#pragma unroll
        for (int mt = 0; mt < 4; ++mt)
#pragma unroll
            for (int nt = 0; nt < 4; ++nt)
                acc[mt][nt] = __builtin_amdgcn_wmma_f32_16x16x32_f16(
                    false, af[mt], false, bf[nt],
                    (short)0, acc[mt][nt], false, false);

        // ---- dump staged registers into the other buffer
        if (more) {
#pragma unroll
            for (int i = 0; i < 2; ++i) *(v8h*)&sA[cur ^ 1][aoffL[i]] = ra[i];
#pragma unroll
            for (int i = 0; i < 4; ++i) *(v8h*)&sB[cur ^ 1][boffL[i]] = rb[i];
        }
        __syncthreads();
        cur ^= 1;
    }

    // ---- epilogue: C/D layout => VGPR v holds M = v + 8*hsel, N = r
    _Float16* __restrict__ ob = out + (long)mBase * OUT_F + nBase;  // uniform base
#pragma unroll
    for (int mt = 0; mt < 4; ++mt) {
        const int mrow = mw * 64 + mt * 16 + hsel * 8;
#pragma unroll
        for (int nt = 0; nt < 4; ++nt) {
            const int n = nw * 64 + nt * 16 + r;
#pragma unroll
            for (int v = 0; v < 8; ++v)
                ob[(mrow + v) * OUT_F + n] = (_Float16)acc[mt][nt][v];
        }
    }
}

// ---------------------------------------------------------------------------
// Launcher. Inputs (setup_inputs order): x f32, codes i32, norms f16,
// centroids f32, H f32 (unused: FWHT reproduces the normalized Sylvester H).
// Workspace: [0, 16 MB) fp16 xh;  [16 MB, 106 MB) fp16 W'.
// ---------------------------------------------------------------------------
extern "C" void kernel_launch(void* const* d_in, const int* in_sizes, int n_in,
                              void* d_out, int out_size, void* d_ws, size_t ws_size,
                              hipStream_t stream) {
    const float*    x         = (const float*)d_in[0];
    const int*      codes     = (const int*)d_in[1];
    const _Float16* norms     = (const _Float16*)d_in[2];
    const float*    centroids = (const float*)d_in[3];

    _Float16* xh  = (_Float16*)d_ws;
    _Float16* wq  = (_Float16*)((char*)d_ws + (size_t)M_TOK * IN_F * sizeof(_Float16));
    _Float16* out = (_Float16*)d_out;

    // 1) Hadamard transform of x: 2048*32 = 65536 chunks, 2 per block
    fwht_x_kernel<<<dim3((M_TOK * (IN_F / BS)) / 2), dim3(256), 0, stream>>>(x, xh);

    // 2) Dequant codes -> fp16 W' : 45,088,768 codes / 8 per thread / 256 = 22016 blocks
    decode_w_kernel<<<dim3(((long)OUT_F * IN_F) / (8 * 256)), dim3(256), 0, stream>>>(
        codes, norms, centroids, wq);

    // 3) WMMA GEMM: grid 43 (N) x 16 (M) block tiles of 128x256
    gemm_wmma_kernel<<<dim3(OUT_F / BN, M_TOK / BM), dim3(256), 0, stream>>>(xh, wq, out);
}
